// WaveletResidualQuantizer_80590766342859
// MI455X (gfx1250) — compile-verified, matrix operand order checked by reading, not run on hardware
//
#include <hip/hip_runtime.h>
#include <hip/hip_bf16.h>

typedef __attribute__((ext_vector_type(16))) __bf16 v16bf;
typedef __attribute__((ext_vector_type(8)))  float  v8f;

// Problem constants
#define BATCH   16
#define CHAN    256
#define HI      128
#define WI      128
#define HO      64
#define WO      64
#define NCODE   1024
#define NSUB    4
#define NPTS    65536          // BATCH*HO*WO per subband
#define SUBELEM 16777216u      // NPTS*CHAN
#define OUTQ    16777216       // final_quantized elements
#define POS_LOSS 16777216
#define POS_PERP 16777217
#define POS_IDXF 16777221

#define LDSB_STRIDE 264        // 256 + 8 pad -> conflict-free b128 reads
#define LDSB_ELEMS  (16 * LDSB_STRIDE)

typedef unsigned short __attribute__((address_space(3))) lds_ushort;

__device__ __forceinline__ unsigned short f2bf(float f) {
    unsigned u = __float_as_uint(f);
    u += 0x7FFFu + ((u >> 16) & 1u);   // round-to-nearest-even
    return (unsigned short)(u >> 16);
}

// ---------------------------------------------------------------- prep ------
// One block per code row: f32->bf16 codebook + 0.5*|e|^2.
__global__ void prep_codebook(const float* __restrict__ cb,
                              unsigned short* __restrict__ cbBf,
                              float* __restrict__ esqHalf) {
    __shared__ float red[8];
    int blk = blockIdx.x;               // 0..4095  (s*1024 + code)
    int t = threadIdx.x;                // 0..255 = channel
    float e = cb[(size_t)blk * CHAN + t];
    cbBf[(size_t)blk * CHAN + t] = f2bf(e);
    float ss = e * e;
    #pragma unroll
    for (int m = 1; m < 32; m <<= 1) ss += __shfl_xor(ss, m, 32);
    if ((t & 31) == 0) red[t >> 5] = ss;
    __syncthreads();
    if (t == 0) {
        float tot = 0.f;
        #pragma unroll
        for (int w = 0; w < 8; ++w) tot += red[w];
        esqHalf[blk] = 0.5f * tot;
    }
}

// ----------------------------------------------------------------- dwt ------
// Block tile: fixed (b, i), 32 channels, all 64 output-j.  LDS transpose so
// global reads are j-coalesced and bf16 writes are channel-coalesced.
__global__ void dwt_kernel(const float* __restrict__ x,
                           unsigned short* __restrict__ sub) {
    __shared__ float lds[32 * 261];     // per-channel stride 261 -> conflict-free
    int blk = blockIdx.x;               // 16*64*8 = 8192
    int cb = blk & 7;
    int i  = (blk >> 3) & 63;
    int b  = blk >> 9;
    int t  = threadIdx.x;

    // read 32ch x 2 rows x 128 f32, coalesced over jj
    #pragma unroll
    for (int it = 0; it < 32; ++it) {
        int lin = it * 256 + t;                 // [0,8192)
        int c  = lin >> 8;
        int r  = (lin >> 7) & 1;
        int jj = lin & 127;
        lds[c * 261 + r * 128 + jj] =
            x[(((size_t)(b * CHAN + cb * 32 + c)) * HI + (2 * i + r)) * WI + jj];
    }
    __syncthreads();

    #pragma unroll
    for (int it = 0; it < 8; ++it) {
        int lin = it * 256 + t;                 // [0,2048) = 32c x 64j
        int c = lin & 31;
        int j = lin >> 5;
        const float* p = &lds[c * 261];
        float a = p[2 * j], bb = p[2 * j + 1];
        float cc = p[128 + 2 * j], d = p[128 + 2 * j + 1];
        float ll = (a + bb + cc + d) * 0.5f;
        float lh = (a + bb - cc - d) * 0.5f;
        float hl = (a - bb + cc - d) * 0.5f;
        float hh = (a - bb - cc + d) * 0.5f;
        unsigned n = ((unsigned)(b * HO + i)) * WO + j;
        unsigned base = n * CHAN + (cb * 32 + c);
        sub[0u * SUBELEM + base] = f2bf(ll);
        sub[1u * SUBELEM + base] = f2bf(lh);
        sub[2u * SUBELEM + base] = f2bf(hl);
        sub[3u * SUBELEM + base] = f2bf(hh);
    }
}

// ---------------------------------------------------------- gemm+argmin -----
// 2048 blocks x 256 threads.  Wave = one 16-row block of zf; 64 code tiles,
// 8x v_wmma_f32_16x16x32_bf16 per tile.  B tiles staged into double-buffered
// LDS with CDNA5 async loads (global_load_async_to_lds_b128 / ASYNCcnt),
// overlapping the tile t+1 fill with tile t compute -> one barrier per tile.
union Frag { v16bf v; uint4 q[2]; };

__global__ void gemm_argmin(const unsigned short* __restrict__ subB,
                            const unsigned short* __restrict__ cbBf,
                            const float* __restrict__ esqHalf,
                            int*   __restrict__ idxInt,
                            float* __restrict__ idxOutF,
                            float* __restrict__ counts) {
    __shared__ unsigned short ldsB[2][LDSB_ELEMS];
    __shared__ float ldsEsq[NCODE];

    int s        = blockIdx.x >> 9;             // subband
    int blockRow = blockIdx.x & 511;
    int t    = threadIdx.x;
    int lane = t & 31;
    int wid  = t >> 5;
    int col  = lane & 15;
    int kb   = (lane < 16) ? 0 : 8;             // bf16 A/B lane K-offset
    unsigned rows_base = ((unsigned)(blockRow * 8 + wid)) << 4;
    unsigned row = rows_base + col;

    // stage 0.5*|e|^2 for this subband (visible after first in-loop barrier)
    #pragma unroll
    for (int k = 0; k < 4; ++k)
        ldsEsq[t + 256 * k] = esqHalf[(s << 10) + t + 256 * k];

    // async stage of one 16-code tile into LDS buffer `buf`:
    // 512 x 16B chunks, 2 per thread, VGPR-free global->LDS path
    auto issue_tile = [&](int tile, int buf) {
        #pragma unroll
        for (int h = 0; h < 2; ++h) {
            int ch   = t + h * 256;             // chunk id [0,512)
            int code = ch >> 5;
            int seg  = ch & 31;
            const unsigned short* g =
                cbBf + ((size_t)((s << 10) + (tile << 4) + code)) * CHAN + seg * 8;
            lds_ushort* l = (lds_ushort*)&ldsB[buf][code * LDSB_STRIDE + seg * 8];
            asm volatile("global_load_async_to_lds_b128 %0, %1, off"
                         :: "v"(l), "v"(g) : "memory");
        }
    };

    // preload A fragments (16 rows x 256 K) into VGPRs
    Frag afrag[8];
    {
        const uint4* ap = reinterpret_cast<const uint4*>(
            subB + (size_t)s * SUBELEM + (size_t)row * CHAN + kb);
        #pragma unroll
        for (int ks = 0; ks < 8; ++ks) {
            afrag[ks].q[0] = ap[ks * 4];        // +ks*32 ushorts
            afrag[ks].q[1] = ap[ks * 4 + 2];    // +16 ushorts further
        }
    }

    float bestv[8];
    int   besti[8];
    #pragma unroll
    for (int v = 0; v < 8; ++v) { bestv[v] = -3.0e38f; besti[v] = 0; }

    issue_tile(0, 0);                           // prologue fill

    for (int tile = 0; tile < 64; ++tile) {
        // own async writes for this tile have landed ...
        asm volatile("s_wait_asynccnt 0x0" ::: "memory");
        // ... and after the barrier, everyone's have
        __syncthreads();
        if (tile + 1 < 64) issue_tile(tile + 1, (tile + 1) & 1);  // overlap fill

        const unsigned short* bbuf = &ldsB[tile & 1][0];
        v8f acc = {};
        #pragma unroll
        for (int ks = 0; ks < 8; ++ks) {
            Frag bf;
            const uint4* lp = reinterpret_cast<const uint4*>(
                bbuf + col * LDSB_STRIDE + ks * 32 + kb);
            bf.q[0] = lp[0];
            bf.q[1] = lp[2];
            acc = __builtin_amdgcn_wmma_f32_16x16x32_bf16(
                false, afrag[ks].v, false, bf.v, (short)0, acc, false, false);
        }

        float eh = ldsEsq[(tile << 4) + col];
        int cand = (tile << 4) + col;
        #pragma unroll
        for (int v = 0; v < 8; ++v) {
            float sc = acc[v] - eh;             // z.e - 0.5|e|^2  (maximize)
            if (sc > bestv[v]) { bestv[v] = sc; besti[v] = cand; }
        }
    }

    // 16-lane argmax reduction per row (first-index tie-break == argmin(d) first)
    int half = lane >> 4;
    #pragma unroll
    for (int v = 0; v < 8; ++v) {
        float bv = bestv[v]; int bi = besti[v];
        #pragma unroll
        for (int m = 1; m < 16; m <<= 1) {
            float ov = __shfl_xor(bv, m, 32);
            int   oi = __shfl_xor(bi, m, 32);
            if (ov > bv || (ov == bv && oi < bi)) { bv = ov; bi = oi; }
        }
        if (col == 0) {
            unsigned r = rows_base + v + 8 * half;
            idxInt[(s << 16) + r] = bi;
            idxOutF[(s << 16) + r] = (float)bi;
            atomicAdd(&counts[(s << 10) + bi], 1.0f);
        }
    }
}

// --------------------------------------------------------------- finalq -----
// One thread per output element (B,C,h,w): recompute DWT in f32, gather 4
// codes, write sum, accumulate per-subband SSE into spread buckets.
__global__ void finalq_kernel(const float* __restrict__ x,
                              const float* __restrict__ cb,
                              const int* __restrict__ idxInt,
                              float* __restrict__ out,
                              float* __restrict__ lossBuckets) {
    __shared__ float wsum[8][4];
    unsigned flat = blockIdx.x * 256u + threadIdx.x;   // (B,C,h,w) flat
    int j = flat & 63;
    int i = (flat >> 6) & 63;
    int c = (flat >> 12) & 255;
    int b = flat >> 20;

    const float* xp = x + (((size_t)(b * CHAN + c)) * HI + 2 * i) * WI + 2 * j;
    float a = xp[0], bb = xp[1], cc = xp[WI], d = xp[WI + 1];
    float zp[4];
    zp[0] = (a + bb + cc + d) * 0.5f;
    zp[1] = (a + bb - cc - d) * 0.5f;
    zp[2] = (a - bb + cc - d) * 0.5f;
    zp[3] = (a - bb - cc + d) * 0.5f;

    unsigned n = ((unsigned)(b * HO + i)) * WO + j;
    float sumq = 0.f;
    float sq[4];
    #pragma unroll
    for (int s = 0; s < 4; ++s) {
        int id = idxInt[(s << 16) + n];
        float q = cb[((size_t)((s << 10) + id)) * CHAN + c];
        sumq += q;
        float r = q - zp[s];
        sq[s] = r * r;
    }
    out[flat] = sumq;

    int lane = threadIdx.x & 31, wid = threadIdx.x >> 5;
    #pragma unroll
    for (int s = 0; s < 4; ++s) {
        float v = sq[s];
        #pragma unroll
        for (int m = 1; m < 32; m <<= 1) v += __shfl_xor(v, m, 32);
        if (lane == 0) wsum[wid][s] = v;
    }
    __syncthreads();
    if (threadIdx.x < 4) {
        float tot = 0.f;
        #pragma unroll
        for (int w = 0; w < 8; ++w) tot += wsum[w][threadIdx.x];
        atomicAdd(&lossBuckets[(threadIdx.x << 10) + (blockIdx.x & 1023)], tot);
    }
}

// -------------------------------------------------------------- finalize ----
__global__ void finalize_kernel(const float* __restrict__ counts,
                                const float* __restrict__ lossBuckets,
                                float* __restrict__ out) {
    __shared__ float sbuf[1024];
    int t = threadIdx.x;

    // total loss = 1.25 * (sum of all SSE) / 2^24
    float v = lossBuckets[t] + lossBuckets[1024 + t] +
              lossBuckets[2048 + t] + lossBuckets[3072 + t];
    sbuf[t] = v;
    for (int off = 512; off > 0; off >>= 1) {
        __syncthreads();
        if (t < off) sbuf[t] += sbuf[t + off];
    }
    __syncthreads();
    if (t == 0) out[POS_LOSS] = 1.25f * sbuf[0] * (1.0f / 16777216.0f);

    // perplexities
    for (int s = 0; s < 4; ++s) {
        __syncthreads();
        float p = counts[(s << 10) + t] * (1.0f / 65536.0f);
        sbuf[t] = p * __logf(p + 1e-10f);
        for (int off = 512; off > 0; off >>= 1) {
            __syncthreads();
            if (t < off) sbuf[t] += sbuf[t + off];
        }
        __syncthreads();
        if (t == 0) out[POS_PERP + s] = __expf(-sbuf[0]);
    }
}

// ---------------------------------------------------------------------------
extern "C" void kernel_launch(void* const* d_in, const int* in_sizes, int n_in,
                              void* d_out, int out_size, void* d_ws, size_t ws_size,
                              hipStream_t stream) {
    const float* x  = (const float*)d_in[0];
    const float* cb = (const float*)d_in[1];
    float* out = (float*)d_out;

    char* ws = (char*)d_ws;
    unsigned short* subBf   = (unsigned short*)(ws);                      // 134,217,728 B
    unsigned short* cbBf    = (unsigned short*)(ws + 134217728u);         //   2,097,152 B
    float*          esqHalf = (float*)(ws + 136314880u);                  //      16,384 B
    int*            idxInt  = (int*)  (ws + 136331264u);                  //   1,048,576 B
    float*          counts  = (float*)(ws + 137379840u);                  //      16,384 B
    float*          lossBk  = (float*)(ws + 137396224u);                  //      16,384 B

    // zero counts + loss buckets (contiguous 32 KB)
    hipMemsetAsync(counts, 0, 32768, stream);

    prep_codebook<<<NSUB * NCODE, 256, 0, stream>>>(cb, cbBf, esqHalf);
    dwt_kernel<<<BATCH * HO * 8, 256, 0, stream>>>(x, subBf);
    gemm_argmin<<<2048, 256, 0, stream>>>(subBf, cbBf, esqHalf,
                                          idxInt, out + POS_IDXF, counts);
    finalq_kernel<<<OUTQ / 256, 256, 0, stream>>>(x, cb, idxInt, out, lossBk);
    finalize_kernel<<<1, 1024, 0, stream>>>(counts, lossBk, out);
}
